// QuadraticFormModel_4458176053374
// MI455X (gfx1250) — matrix operand order repaired and checked by reading, not compiled, not run on hardware
//
#include <hip/hip_runtime.h>

// MI455X / gfx1250: wave32, WMMA bf16 16x16x32, f32 accumulate.
// out[i] = sum_j x[i,j]*(x@A)[i,j] + x[i].v + b
// Split-bf16 (hi/lo) for near-fp32 accuracy at bf16-WMMA speed.

typedef __attribute__((ext_vector_type(16))) __bf16 v16bf;
typedef __attribute__((ext_vector_type(8)))  float  v8f;

#define DD 256

// ---------------------------------------------------------------------------
// Prep: split A (DxD f32) into bf16 hi/lo, pre-swizzled into the WMMA
// B-fragment layout so the hot loop does contiguous 32B loads per lane.
// B-matrix 32x16 bf16 order used here (contiguous k, not AMD's table order): VGPR j holds K=2j,2j+1 for
// lanes 0-15 and K=16+2j,16+2j+1 for lanes 16-31, N = lane%16.
// Element e (0..15) of the packed fragment <-> K = k0 + 16*half + e.
// Fragment base index: ((chunk*16 + ntile)*32 + lane)*16
// ---------------------------------------------------------------------------
__global__ __launch_bounds__(256) void qf_prep_w(const float* __restrict__ W,
                                                 __bf16* __restrict__ wh,
                                                 __bf16* __restrict__ wl) {
  int idx = blockIdx.x * 256 + threadIdx.x;   // 65536 elements of A
  int r   = idx >> 8;                         // K index (row of A)
  int col = idx & 255;                        // N index (col of A)
  float wv = W[idx];
  __bf16 h = (__bf16)wv;
  __bf16 l = (__bf16)(wv - (float)h);
  int c    = r >> 5;          // K chunk of 32
  int half = (r >> 4) & 1;    // upper/lower 16 of the chunk
  int e    = r & 15;          // element within fragment
  int t    = col >> 4;        // N tile
  int lh   = col & 15;        // lane within half-wave
  int lane = half * 16 + lh;
  int dst  = ((c * 16 + t) * 32 + lane) * 16 + e;
  wh[dst] = h;
  wl[dst] = l;
}

// ---------------------------------------------------------------------------
// Main: one wave32 per 16-row tile of x. 8 waves / 256-thread block.
// ---------------------------------------------------------------------------
__global__ __launch_bounds__(256) void qf_main(const float* __restrict__ X,
                                               const float* __restrict__ V,
                                               const float* __restrict__ Bias,
                                               const __bf16* __restrict__ wh,
                                               const __bf16* __restrict__ wl,
                                               float* __restrict__ out) {
  const int wave = threadIdx.x >> 5;
  const int lane = threadIdx.x & 31;
  const int half = lane >> 4;        // 0: lanes 0-15, 1: lanes 16-31
  const int lh   = lane & 15;
  const int row0 = (blockIdx.x * 8 + wave) * 16;
  const float* xrow = X + (size_t)(row0 + lh) * DD;

  // Build 8 hi/lo A-fragments (16-bit A 16x32, contiguous k order matching the B packing):
  // lane (half,lh) holds row M=lh, K = k0 + 16*half + {0..15}.
  v16bf ahi[8], alo[8];
#pragma unroll
  for (int c = 0; c < 8; ++c) {
#pragma unroll
    for (int e = 0; e < 16; ++e) {
      float xv = xrow[c * 32 + half * 16 + e];
      __bf16 h = (__bf16)xv;
      ahi[c][e] = h;
      alo[c][e] = (__bf16)(xv - (float)h);
    }
  }

  // rowsum[r]: partial sum_j Y[m,j]*X[m,j] for m = row0 + r + 8*half,
  // columns n = t*16 + lh (reduced across lanes at the end).
  v8f rowsum = {};
  for (int t = 0; t < 16; ++t) {
    v8f acc = {};
#pragma unroll
    for (int c = 0; c < 8; ++c) {
      const size_t fb = ((size_t)(c * 16 + t) * 32 + lane) * 16;
      const v16bf bh = *(const v16bf*)(wh + fb);
      const v16bf bl = *(const v16bf*)(wl + fb);
      acc = __builtin_amdgcn_wmma_f32_16x16x32_bf16(false, ahi[c], false, bh,
                                                    (short)0, acc, false, false);
      acc = __builtin_amdgcn_wmma_f32_16x16x32_bf16(false, ahi[c], false, bl,
                                                    (short)0, acc, false, false);
      acc = __builtin_amdgcn_wmma_f32_16x16x32_bf16(false, alo[c], false, bh,
                                                    (short)0, acc, false, false);
    }
    // Fuse elementwise multiply with x on the C/D register layout:
    // acc[r] @ lane (half,lh) = Y[row0 + r + 8*half][t*16 + lh]
#pragma unroll
    for (int r = 0; r < 8; ++r) {
      float xe = X[(size_t)(row0 + 8 * half + r) * DD + t * 16 + lh];
      rowsum[r] += acc[r] * xe;
    }
  }

  // Linear term in pure f32: lane (half,lh) sums K in [128*half, 128*half+128)
  // for row row0+lh; xor-16 combine gives the full dot product.
  float lin = 0.f;
  const int kb = half * 128;
#pragma unroll 8
  for (int k = 0; k < 128; ++k) lin += xrow[kb + k] * V[kb + k];
  lin += __shfl_xor(lin, 16);
  const float bias = Bias[0];

  // Butterfly-reduce rowsum across each 16-lane group, then write 16 rows.
#pragma unroll
  for (int r = 0; r < 8; ++r) {
    float s = rowsum[r];
    s += __shfl_xor(s, 1);
    s += __shfl_xor(s, 2);
    s += __shfl_xor(s, 4);
    s += __shfl_xor(s, 8);
    float hi = __shfl_xor(s, 16);        // upper group's sum (rows row0+8+r)
    if (lane == r)     out[row0 + r]     = s  + lin + bias;  // lin@lane r   = row r
    if (lane == 8 + r) out[row0 + 8 + r] = hi + lin + bias;  // lin@lane 8+r = row 8+r
  }
}

// ---------------------------------------------------------------------------
// Launch. d_in: x[B*256], A[256*256], v[256], b[1]. d_out: B floats.
// d_ws: needs 2 * 256*256 * 2B = 256 KB for pre-split A.
// ---------------------------------------------------------------------------
extern "C" void kernel_launch(void* const* d_in, const int* in_sizes, int n_in,
                              void* d_out, int out_size, void* d_ws, size_t ws_size,
                              hipStream_t stream) {
  const float* x = (const float*)d_in[0];
  const float* A = (const float*)d_in[1];
  const float* v = (const float*)d_in[2];
  const float* b = (const float*)d_in[3];
  float* out = (float*)d_out;

  __bf16* wh = (__bf16*)d_ws;
  __bf16* wl = wh + DD * DD;

  const int rows = in_sizes[0] / DD;       // B = 131072

  qf_prep_w<<<(DD * DD) / 256, 256, 0, stream>>>(A, wh, wl);
  qf_main<<<rows / 128, 256, 0, stream>>>(x, v, b, wh, wl, out);
}
